// LinkPredModel_39737037422592
// MI455X (gfx1250) — compile-verified, matrix operand order checked by reading, not run on hardware
//
#include <hip/hip_runtime.h>
#include <hip/hip_bf16.h>
#include <stdint.h>

typedef __bf16 bf16_t;
typedef __attribute__((ext_vector_type(16))) __bf16 v16bf;
typedef __attribute__((ext_vector_type(8)))  __bf16 v8bf;
typedef __attribute__((ext_vector_type(8)))  float  v8f;
typedef int v4i __attribute__((vector_size(16)));   // b128 payload type for async DMA

#define NNODES 100000
#define NEDGES 100000
#define DIM    512

#ifndef __has_builtin
#define __has_builtin(x) 0
#endif

#if __has_builtin(__builtin_amdgcn_global_load_async_to_lds_b128) && \
    __has_builtin(__builtin_amdgcn_s_wait_asynccnt)
#define USE_ASYNC_LDS 1
#else
#define USE_ASYNC_LDS 0
#endif

#define GLOBAL_AS __attribute__((address_space(1)))
#define LDS_AS    __attribute__((address_space(3)))

// ---------------- simple conversion kernels ----------------

__global__ __launch_bounds__(256) void cvt_f32_bf16(const float* __restrict__ in,
                                                    bf16_t* __restrict__ out, int n) {
  int i = blockIdx.x * 256 + threadIdx.x;
  if (i < n) out[i] = (bf16_t)in[i];
}

// in: [K,N] f32 row-major  ->  out: [N,K] bf16 row-major (transposed)
__global__ __launch_bounds__(256) void transpose_cvt(const float* __restrict__ in,
                                                     bf16_t* __restrict__ out,
                                                     int K, int N) {
  int i = blockIdx.x * 256 + threadIdx.x;
  if (i < K * N) {
    int n = i / K;
    int k = i - n * K;
    out[i] = (bf16_t)in[(size_t)k * N + n];
  }
}

// z[e] = concat(h[src[e]], h[dst[e]])  (bf16, DIM each side)
__global__ __launch_bounds__(256) void gather_concat(const bf16_t* __restrict__ h,
                                                     const int* __restrict__ src,
                                                     const int* __restrict__ dst,
                                                     bf16_t* __restrict__ z, int E) {
  int e = blockIdx.x;
  int t = threadIdx.x;
  if (e >= E) return;
  int row = (t < 128) ? src[e] : dst[e];
  int c = (t & 127) * 4;                       // 4 bf16 = 8 bytes per thread
  const uint64_t* s = (const uint64_t*)(h + (size_t)row * DIM + c);
  uint64_t* d = (uint64_t*)(z + (size_t)e * (2 * DIM) + ((t < 128) ? 0 : DIM) + c);
  *d = *s;
}

// ---------------- fused WMMA GEMM: C = relu(A*B^T + bias) [+ A] ----------------
// A:  [M,K] bf16 row-major, lda = K
// Bt: [N,K] bf16 row-major (weights pre-transposed), ldb = K
// C:  [M,N] bf16 row-major
// Block: 256 thr = 8 waves, tile 128(M) x 128(N), K step 32.
// Per-wave tile: 64(M) x 32(N) = 4x2 wmma 16x16 accumulators.
// Staging: async global->LDS DMA (ASYNCcnt) with LDS double buffering when
// available; otherwise batched register loads -> LDS stores.

#define LDS_STRIDE 40   // 32 + 8 pad -> 80B rows, 16B aligned, bank-conflict free

__global__ __launch_bounds__(256) void gemm_bf16_wmma(
    const bf16_t* __restrict__ A,
    const bf16_t* __restrict__ Bt,
    const float*  __restrict__ bias,
    bf16_t* __restrict__ C,
    int M, int N, int K, int add_residual)
{
#if USE_ASYNC_LDS
  __shared__ bf16_t sA[2][128 * LDS_STRIDE];
  __shared__ bf16_t sB[2][128 * LDS_STRIDE];
#else
  __shared__ bf16_t sA[1][128 * LDS_STRIDE];
  __shared__ bf16_t sB[1][128 * LDS_STRIDE];
#endif

  const int tid   = threadIdx.x;
  const int lane  = tid & 31;
  const int wid   = tid >> 5;       // 0..7
  const int waveM = wid & 1;        // 2 waves along M (64 rows each)
  const int waveN = wid >> 1;       // 4 waves along N (32 cols each)
  const int lr    = lane & 15;
  const int lh    = lane >> 4;      // K-half select per ISA 16-bit A/B layout
  const int m0    = blockIdx.x * 128;
  const int n0    = blockIdx.y * 128;

  // K-invariant staging coordinates: 512 chunks of 8 bf16; 2 chunks/thread/tile
  const int c0 = tid, c1 = tid + 256;
  const int r0 = c0 >> 2, col0 = (c0 & 3) * 8;
  const int r1 = c1 >> 2, col1 = (c1 & 3) * 8;
  int gr0 = m0 + r0; if (gr0 >= M) gr0 = M - 1;   // clamp keeps EXEC uniform
  int gr1 = m0 + r1; if (gr1 >= M) gr1 = M - 1;
  const int nr0 = n0 + r0, nr1 = n0 + r1;         // N always multiple of 128

  v8f acc[4][2] = {};

#if USE_ASYNC_LDS
  auto stage = [&](int p, int k0) {
    __builtin_amdgcn_global_load_async_to_lds_b128(
        (GLOBAL_AS v4i*)(A + (size_t)gr0 * K + k0 + col0),
        (LDS_AS v4i*)&sA[p][r0 * LDS_STRIDE + col0], 0, 0);
    __builtin_amdgcn_global_load_async_to_lds_b128(
        (GLOBAL_AS v4i*)(A + (size_t)gr1 * K + k0 + col1),
        (LDS_AS v4i*)&sA[p][r1 * LDS_STRIDE + col1], 0, 0);
    __builtin_amdgcn_global_load_async_to_lds_b128(
        (GLOBAL_AS v4i*)(Bt + (size_t)nr0 * K + k0 + col0),
        (LDS_AS v4i*)&sB[p][r0 * LDS_STRIDE + col0], 0, 0);
    __builtin_amdgcn_global_load_async_to_lds_b128(
        (GLOBAL_AS v4i*)(Bt + (size_t)nr1 * K + k0 + col1),
        (LDS_AS v4i*)&sB[p][r1 * LDS_STRIDE + col1], 0, 0);
  };

  stage(0, 0);
  __builtin_amdgcn_s_wait_asynccnt(0);
  __syncthreads();
  int p = 0;
#endif

  for (int k0 = 0; k0 < K; k0 += 32) {
#if USE_ASYNC_LDS
    if (k0 + 32 < K) stage(p ^ 1, k0 + 32);   // DMA next tile during compute
    const bf16_t* lA = sA[p];
    const bf16_t* lB = sB[p];
#else
    // batched: 4 independent global loads -> one wait -> 4 LDS stores
    v8bf ra0 = *(const v8bf*)(A  + (size_t)gr0 * K + k0 + col0);
    v8bf ra1 = *(const v8bf*)(A  + (size_t)gr1 * K + k0 + col1);
    v8bf rb0 = *(const v8bf*)(Bt + (size_t)nr0 * K + k0 + col0);
    v8bf rb1 = *(const v8bf*)(Bt + (size_t)nr1 * K + k0 + col1);
    *(v8bf*)&sA[0][r0 * LDS_STRIDE + col0] = ra0;
    *(v8bf*)&sA[0][r1 * LDS_STRIDE + col1] = ra1;
    *(v8bf*)&sB[0][r0 * LDS_STRIDE + col0] = rb0;
    *(v8bf*)&sB[0][r1 * LDS_STRIDE + col1] = rb1;
    __syncthreads();
    if (k0 + 32 < K)
      __builtin_prefetch(A + (size_t)gr0 * K + k0 + 32 + col0, 0, 1);
    const bf16_t* lA = sA[0];
    const bf16_t* lB = sB[0];
#endif

    // ---- fragments (ISA 16-bit 16x32 layout: lanes0-15 K{0..7,16..23},
    //      lanes16-31 K{8..15,24..31}); B symmetric since stored transposed.
    v16bf afrag[4], bfrag[2];
    #pragma unroll
    for (int mi = 0; mi < 4; mi++) {
      int r = waveM * 64 + mi * 16 + lr;
      v8bf lo = *(const v8bf*)&lA[r * LDS_STRIDE + lh * 8];
      v8bf hi = *(const v8bf*)&lA[r * LDS_STRIDE + 16 + lh * 8];
      afrag[mi] = __builtin_shufflevector(lo, hi, 0,1,2,3,4,5,6,7,8,9,10,11,12,13,14,15);
    }
    #pragma unroll
    for (int ni = 0; ni < 2; ni++) {
      int r = waveN * 32 + ni * 16 + lr;
      v8bf lo = *(const v8bf*)&lB[r * LDS_STRIDE + lh * 8];
      v8bf hi = *(const v8bf*)&lB[r * LDS_STRIDE + 16 + lh * 8];
      bfrag[ni] = __builtin_shufflevector(lo, hi, 0,1,2,3,4,5,6,7,8,9,10,11,12,13,14,15);
    }

    #pragma unroll
    for (int mi = 0; mi < 4; mi++)
      #pragma unroll
      for (int ni = 0; ni < 2; ni++)
        acc[mi][ni] = __builtin_amdgcn_wmma_f32_16x16x32_bf16(
            false, afrag[mi], false, bfrag[ni], (short)0, acc[mi][ni], false, false);

#if USE_ASYNC_LDS
    __builtin_amdgcn_s_wait_asynccnt(0);  // next tile landed in LDS
    __syncthreads();                      // all waves done reading buf[p]
    p ^= 1;
#else
    __syncthreads();
#endif
  }

  // ---- epilogue: bias + relu (+ residual = A input) -> bf16
  #pragma unroll
  for (int mi = 0; mi < 4; mi++) {
    #pragma unroll
    for (int ni = 0; ni < 2; ni++) {
      int col = n0 + waveN * 32 + ni * 16 + lr;
      float bv = bias[col];
      #pragma unroll
      for (int r = 0; r < 8; r++) {
        int row = m0 + waveM * 64 + mi * 16 + lh * 8 + r;   // ISA C/D layout
        if (row < M) {
          float v = acc[mi][ni][r] + bv;
          v = fmaxf(v, 0.0f);
          if (add_residual) v += (float)A[(size_t)row * K + col];
          C[(size_t)row * N + col] = (bf16_t)v;
        }
      }
    }
  }
}

// ---------------- final 512 -> 1 GEMV (wave32 dot + shuffle reduce) ----------------
__global__ __launch_bounds__(256) void score_kernel(const bf16_t* __restrict__ z,
                                                    const float* __restrict__ w,
                                                    const float* __restrict__ b,
                                                    float* __restrict__ out, int E) {
  int g = threadIdx.x >> 5, lane = threadIdx.x & 31;
  int e = blockIdx.x * 8 + g;
  if (e >= E) return;
  const bf16_t* zr = z + (size_t)e * DIM;
  float s = 0.0f;
  #pragma unroll 4
  for (int j = lane; j < DIM; j += 32) s += (float)zr[j] * w[j];
  #pragma unroll
  for (int o = 16; o > 0; o >>= 1) s += __shfl_xor(s, o, 32);
  if (lane == 0) out[e] = s + b[0];
}

// ---------------- launcher ----------------
extern "C" void kernel_launch(void* const* d_in, const int* in_sizes, int n_in,
                              void* d_out, int out_size, void* d_ws, size_t ws_size,
                              hipStream_t stream) {
  const float* x       = (const float*)d_in[0];
  const int*   pos_src = (const int*)d_in[1];
  const int*   pos_dst = (const int*)d_in[2];
  const int*   neg_src = (const int*)d_in[3];
  const int*   neg_dst = (const int*)d_in[4];
  const float* W1 = (const float*)d_in[5];  const float* b1 = (const float*)d_in[6];
  const float* W2 = (const float*)d_in[7];  const float* b2 = (const float*)d_in[8];
  const float* Wo = (const float*)d_in[9];  const float* bo = (const float*)d_in[10];
  const float* P1w = (const float*)d_in[11]; const float* P1b = (const float*)d_in[12];
  const float* P2w = (const float*)d_in[13]; const float* P2b = (const float*)d_in[14];
  const float* P3w = (const float*)d_in[15]; const float* P3b = (const float*)d_in[16];
  float* out = (float*)d_out;

  char* ws = (char*)d_ws;
  size_t off = 0;
  auto carve = [&](size_t bytes) -> void* {
    void* p = ws + off;
    off += (bytes + 255) & ~(size_t)255;
    return p;
  };
  bf16_t* bufA = (bf16_t*)carve((size_t)NNODES * DIM * 2);       // ping
  bf16_t* bufB = (bf16_t*)carve((size_t)NNODES * DIM * 2);       // pong
  bf16_t* bufC = (bf16_t*)carve((size_t)NNODES * 2 * DIM * 2);   // concat / z2
  bf16_t* w1t  = (bf16_t*)carve((size_t)DIM * DIM * 2);
  bf16_t* w2t  = (bf16_t*)carve((size_t)DIM * DIM * 2);
  bf16_t* wot  = (bf16_t*)carve((size_t)DIM * DIM * 2);
  bf16_t* p1t  = (bf16_t*)carve((size_t)DIM * 2 * DIM * 2);
  bf16_t* p2t  = (bf16_t*)carve((size_t)DIM * DIM * 2);

  dim3 blk(256);

  // precision conversion + weight transposes
  cvt_f32_bf16<<<(NNODES * DIM + 255) / 256, blk, 0, stream>>>(x, bufA, NNODES * DIM);
  transpose_cvt<<<(DIM * DIM + 255) / 256, blk, 0, stream>>>(W1, w1t, DIM, DIM);
  transpose_cvt<<<(DIM * DIM + 255) / 256, blk, 0, stream>>>(W2, w2t, DIM, DIM);
  transpose_cvt<<<(DIM * DIM + 255) / 256, blk, 0, stream>>>(Wo, wot, DIM, DIM);
  transpose_cvt<<<(2 * DIM * DIM + 255) / 256, blk, 0, stream>>>(P1w, p1t, 2 * DIM, DIM);
  transpose_cvt<<<(DIM * DIM + 255) / 256, blk, 0, stream>>>(P2w, p2t, DIM, DIM);

  // encoder: h1 = relu(x W1 + b1); h2 = relu(h1 W2 + b2) + h1; h = relu(h2 Wo + bo) + h2
  dim3 ggrid((NNODES + 127) / 128, DIM / 128);
  gemm_bf16_wmma<<<ggrid, blk, 0, stream>>>(bufA, w1t, b1, bufB, NNODES, DIM, DIM, 0);
  gemm_bf16_wmma<<<ggrid, blk, 0, stream>>>(bufB, w2t, b2, bufA, NNODES, DIM, DIM, 1);
  gemm_bf16_wmma<<<ggrid, blk, 0, stream>>>(bufA, wot, bo, bufB, NNODES, DIM, DIM, 1);
  // h lives in bufB

  const int* srcs[2] = {pos_src, neg_src};
  const int* dsts[2] = {pos_dst, neg_dst};
  dim3 egrid((NEDGES + 127) / 128, DIM / 128);
  for (int s = 0; s < 2; s++) {
    gather_concat<<<NEDGES, blk, 0, stream>>>(bufB, srcs[s], dsts[s], bufC, NEDGES);
    gemm_bf16_wmma<<<egrid, blk, 0, stream>>>(bufC, p1t, P1b, bufA, NEDGES, DIM, 2 * DIM, 0);
    gemm_bf16_wmma<<<egrid, blk, 0, stream>>>(bufA, p2t, P2b, bufC, NEDGES, DIM, DIM, 0);
    score_kernel<<<(NEDGES + 7) / 8, blk, 0, stream>>>(bufC, P3w, P3b, out + s * NEDGES, NEDGES);
  }
}
